// self_dilating_Pooling_30605936951856
// MI455X (gfx1250) — matrix operand order repaired and checked by reading, not compiled, not run on hardware
//
#include <hip/hip_runtime.h>
#include <hip/hip_bf16.h>

typedef _Float16 v16h __attribute__((ext_vector_type(16)));
typedef float    v8f  __attribute__((ext_vector_type(8)));
typedef unsigned int v4u __attribute__((ext_vector_type(4)));
typedef int      v8i  __attribute__((ext_vector_type(8)));
typedef int      v4i  __attribute__((ext_vector_type(4)));

#define B_    8
#define C_    512
#define HW_   4096           // 64*64 per plane
#define LDSW  65             // padded row stride: bank = (y+x)%64, conflict-free in all passes

#if __has_builtin(__builtin_amdgcn_tensor_load_to_lds) && __has_builtin(__builtin_amdgcn_s_wait_tensorcnt)
#define HAS_TDM 1
#else
#define HAS_TDM 0
#endif

// ============================================================================
// Kernel 1: global average pool.  gap[b*512+c] = mean(x[b,c,:,:])
// ============================================================================
__global__ void gap_kernel(const float* __restrict__ x, float* __restrict__ gap) {
    __shared__ float red[256];
    const int bc  = blockIdx.x;
    const int tid = threadIdx.x;
    const float4* px = (const float4*)(x + (size_t)bc * HW_);
    float s = 0.f;
#pragma unroll
    for (int k = 0; k < 4; ++k) {
        float4 v = px[tid + k * 256];
        s += v.x + v.y + v.z + v.w;
    }
    red[tid] = s;
    __syncthreads();
#pragma unroll
    for (int off = 128; off > 0; off >>= 1) {
        if (tid < off) red[tid] += red[tid + off];
        __syncthreads();
    }
    if (tid == 0) gap[bc] = red[0] * (1.0f / (float)HW_);
}

// ============================================================================
// Kernel 2: SE bottleneck + routing, one wave (32 lanes), WMMA f32<=f16.
// D rows 8..15 are never read, so lanes 8..15 duplicate rows 0..7 (m&7) —
// no exec-mask predication on the A-fragment loads (lets loads vectorize).
// ============================================================================
__global__ void se_route_kernel(const float* __restrict__ gap,
                                const float* __restrict__ w1,
                                const float* __restrict__ bias1,
                                const float* __restrict__ w2,
                                const float* __restrict__ bias2,
                                int*   __restrict__ qsOut,
                                float* __restrict__ fbOut) {
    __shared__ _Float16 hmid[8 * 32];

    const int lane = threadIdx.x & 31;
    const int m    = lane & 15;
    const int mr   = m & 7;              // duplicated row for lanes with m>=8
    const int hi   = lane >> 4;          // 0 or 1
    const int kbA  = hi * 8;             // unused; the A frag also uses kbB
    const int kbB  = hi * 16;            // B-frag K base within chunk

    // ---- matmul 1: hmid = relu(gap[8,512] @ w1^T[512,32] + b1) ----
    v8f acc0 = {0.f, 0.f, 0.f, 0.f, 0.f, 0.f, 0.f, 0.f};
    v8f acc1 = acc0;
    for (int kc = 0; kc < 16; ++kc) {
        const int k0 = kc * 32;
        v16h a;
#pragma unroll
        for (int i = 0; i < 8; ++i) {
            a[i]     = (_Float16)gap[mr * C_ + k0 + kbB + i];
            a[8 + i] = (_Float16)gap[mr * C_ + k0 + kbB + 8 + i];
        }
        v16h bf0, bf1;
#pragma unroll
        for (int i = 0; i < 16; ++i) {
            bf0[i] = (_Float16)w1[(m)      * C_ + k0 + kbB + i];
            bf1[i] = (_Float16)w1[(m + 16) * C_ + k0 + kbB + i];
        }
        acc0 = __builtin_amdgcn_wmma_f32_16x16x32_f16(false, a, false, bf0, (short)0, acc0, false, false);
        acc1 = __builtin_amdgcn_wmma_f32_16x16x32_f16(false, a, false, bf1, (short)0, acc1, false, false);
    }
    if (lane < 16) {
#pragma unroll
        for (int r = 0; r < 8; ++r) {
            float h0 = acc0[r] + bias1[lane];
            float h1 = acc1[r] + bias1[16 + lane];
            hmid[r * 32 + lane]      = (_Float16)fmaxf(h0, 0.f);
            hmid[r * 32 + 16 + lane] = (_Float16)fmaxf(h1, 0.f);
        }
    }
    __syncthreads();

    // ---- matmul 2: score = hmid[8,32] @ w2^T[32,512] + b2, fused routing ----
    v16h a2;
#pragma unroll
    for (int i = 0; i < 8; ++i) {
        a2[i]     = hmid[mr * 32 + kbB + i];
        a2[8 + i] = hmid[mr * 32 + kbB + 8 + i];
    }
    for (int nt = 0; nt < 32; ++nt) {
        const int n = nt * 16 + m;
        v16h bf;
#pragma unroll
        for (int i = 0; i < 16; ++i)
            bf[i] = (_Float16)w2[n * 32 + kbB + i];
        v8f accS = {0.f, 0.f, 0.f, 0.f, 0.f, 0.f, 0.f, 0.f};
        accS = __builtin_amdgcn_wmma_f32_16x16x32_f16(false, a2, false, bf, (short)0, accS, false, false);
        if (lane < 16) {
#pragma unroll
            for (int r = 0; r < 8; ++r) {
                float s = fmaxf(accS[r] + bias2[n], 0.f);
                int   q = (int)fminf(floorf(s), 5.0f);   // s>=0 so floor>=0
                qsOut[r * C_ + n] = q;
                fbOut[r * C_ + n] = s - (float)q;
            }
        }
    }
}

// ============================================================================
// Kernel 3: routed blended max-pool + residual.  One block per (b,c) plane.
// pool_small (radius R = q_s): separable row/col max in LDS.
// pool_big (radius R+1) == 3x3 max of pool_small (stride-1 SAME nesting).
// out = fb*pool_big + (1-fb)*pool_small + x
// ============================================================================
template <int R>
__device__ __forceinline__ void pool_body(const float* __restrict__ xs,
                                          float* __restrict__ hs,
                                          float* __restrict__ ps,
                                          float* __restrict__ outp,
                                          float fb, float fs, int tid) {
    // ---- horizontal max, radius R: thread owns 16 px of one row ----
    {
        const int y  = tid >> 2;
        const int x0 = (tid & 3) << 4;
        float buf[16 + 2 * R];
#pragma unroll
        for (int i = 0; i < 16 + 2 * R; ++i) {
            int xx = x0 - R + i;
            xx = xx < 0 ? 0 : (xx > 63 ? 63 : xx);
            buf[i] = xs[y * LDSW + xx];
        }
#pragma unroll
        for (int j = 0; j < 16; ++j) {
            float mx = buf[j];
#pragma unroll
            for (int d = 1; d <= 2 * R; ++d) mx = fmaxf(mx, buf[j + d]);
            hs[y * LDSW + x0 + j] = mx;
        }
    }
    __syncthreads();
    // ---- vertical max, radius R: thread owns 16 px of one column ----
    {
        const int x  = tid & 63;
        const int y0 = (tid >> 6) << 4;
        float buf[16 + 2 * R];
#pragma unroll
        for (int i = 0; i < 16 + 2 * R; ++i) {
            int yy = y0 - R + i;
            yy = yy < 0 ? 0 : (yy > 63 ? 63 : yy);
            buf[i] = hs[yy * LDSW + x];
        }
#pragma unroll
        for (int j = 0; j < 16; ++j) {
            float mx = buf[j];
#pragma unroll
            for (int d = 1; d <= 2 * R; ++d) mx = fmaxf(mx, buf[j + d]);
            ps[(y0 + j) * LDSW + x] = mx;
        }
    }
    __syncthreads();
    // ---- pool_big = 3x3 max of ps, blend, residual, coalesced store ----
    {
        const int x  = tid & 63;
        const int y0 = (tid >> 6) << 4;
        const int xl = (x == 0) ? 0 : x - 1;
        const int xr = (x == 63) ? 63 : x + 1;
        float rm[18];
#pragma unroll
        for (int i = 0; i < 18; ++i) {
            int yy = y0 - 1 + i;
            yy = yy < 0 ? 0 : (yy > 63 ? 63 : yy);
            rm[i] = fmaxf(fmaxf(ps[yy * LDSW + xl], ps[yy * LDSW + x]), ps[yy * LDSW + xr]);
        }
#pragma unroll
        for (int j = 0; j < 16; ++j) {
            const int y  = y0 + j;
            float pb = fmaxf(fmaxf(rm[j], rm[j + 1]), rm[j + 2]);
            float pv = ps[y * LDSW + x];
            outp[y * 64 + x] = fb * pb + fs * pv + xs[y * LDSW + x];
        }
    }
}

__global__ void __launch_bounds__(256)
routed_pool_kernel(const float* __restrict__ x,
                   const int*   __restrict__ qsArr,
                   const float* __restrict__ fbArr,
                   float* __restrict__ out) {
    __shared__ float xs[64 * LDSW];
    __shared__ float hs[64 * LDSW];
    __shared__ float ps[64 * LDSW];

    const int bc  = blockIdx.x;
    const int tid = threadIdx.x;
    const int   q  = qsArr[bc];
    const float fb = fbArr[bc];
    const float fs = 1.0f - fb;

#if HAS_TDM
    // ---- Tensor Data Mover: DMA the 64x64 f32 plane global -> LDS.
    // D# per CDNA5 ISA: 2-D tensor, data_size=4B,
    // dims/tile = 64x64, stride 64; LDS padding 1 DWORD per 64 DWORDs
    // (pad_interval=5 -> 2^6=64 DWORDs, pad_amount=0 -> 1 DWORD) => row
    // stride 65 floats, matching LDSW.
    if (tid < 32) {
        const unsigned long long ga = (unsigned long long)(uintptr_t)(x + (size_t)bc * HW_);
        const unsigned int lds_off  = (unsigned int)(uintptr_t)(void*)xs;  // LDS aperture: low 32 bits = LDS offset
        v4u g0;
        g0[0] = 1u;                                        // count=1, user descriptor
        g0[1] = lds_off;                                   // lds_addr [63:32]
        g0[2] = (unsigned int)(ga & 0xFFFFFFFFu);          // global_addr lo
        g0[3] = (unsigned int)((ga >> 32) & 0x01FFFFFFu) | (2u << 30);  // global_addr hi | type=2
        v8i g1;
        g1[0] = (int)((2u << 16) | (1u << 20) | (5u << 22) | (0u << 25)); // data_size=4B, pad_enable, interval=64dw, amount=1dw
        g1[1] = (int)(64u << 16);                          // tensor_dim0 = 64 (bits 79:48 -> low16 here)
        g1[2] = (int)(64u << 16);                          // tensor_dim1 = 64 (bits 111:80)
        g1[3] = (int)(64u << 16);                          // tile_dim0 = 64 (bits 127:112)
        g1[4] = (int)(64u);                                // tile_dim1 = 64 (bits 143:128)
        g1[5] = (int)(64u);                                // tensor_dim0_stride = 64
        g1[6] = 0;
        g1[7] = 0;
        v4i g2 = {0, 0, 0, 0};
        v4i g3 = {0, 0, 0, 0};
#if defined(__clang_major__) && (__clang_major__ >= 23)
        v8i g4 = {0, 0, 0, 0, 0, 0, 0, 0};
        __builtin_amdgcn_tensor_load_to_lds(g0, g1, g2, g3, g4, 0);
#else
        __builtin_amdgcn_tensor_load_to_lds(g0, g1, g2, g3, 0);
#endif
        __builtin_amdgcn_s_wait_tensorcnt(0);
    }
#else
    // fallback: float4 coalesced global loads + scalar LDS stores
    const float4* gx4 = (const float4*)(x + (size_t)bc * HW_);
#pragma unroll
    for (int k = 0; k < 4; ++k) {
        const int idx = tid + k * 256;        // float4 index within plane
        float4 v = gx4[idx];
        const int y  = idx >> 4;
        const int xc = (idx & 15) << 2;
        xs[y * LDSW + xc + 0] = v.x;
        xs[y * LDSW + xc + 1] = v.y;
        xs[y * LDSW + xc + 2] = v.z;
        xs[y * LDSW + xc + 3] = v.w;
    }
#endif
    __syncthreads();

    float* outp = out + (size_t)bc * HW_;
    switch (q) {
        case 0:  pool_body<0>(xs, hs, ps, outp, fb, fs, tid); break;
        case 1:  pool_body<1>(xs, hs, ps, outp, fb, fs, tid); break;
        case 2:  pool_body<2>(xs, hs, ps, outp, fb, fs, tid); break;
        case 3:  pool_body<3>(xs, hs, ps, outp, fb, fs, tid); break;
        case 4:  pool_body<4>(xs, hs, ps, outp, fb, fs, tid); break;
        default: pool_body<5>(xs, hs, ps, outp, fb, fs, tid); break;
    }
}

// ============================================================================
extern "C" void kernel_launch(void* const* d_in, const int* in_sizes, int n_in,
                              void* d_out, int out_size, void* d_ws, size_t ws_size,
                              hipStream_t stream) {
    const float* x  = (const float*)d_in[0];
    const float* w1 = (const float*)d_in[1];
    const float* b1 = (const float*)d_in[2];
    const float* w2 = (const float*)d_in[3];
    const float* b2 = (const float*)d_in[4];
    float* out = (float*)d_out;

    char* ws = (char*)d_ws;
    float* gap = (float*)(ws);                 // 4096 floats
    int*   qs  = (int*)  (ws + 16384);         // 4096 ints
    float* fb  = (float*)(ws + 32768);         // 4096 floats

    gap_kernel<<<dim3(B_ * C_), dim3(256), 0, stream>>>(x, gap);
    se_route_kernel<<<dim3(1), dim3(32), 0, stream>>>(gap, w1, b1, w2, b2, qs, fb);
    routed_pool_kernel<<<dim3(B_ * C_), dim3(256), 0, stream>>>(x, qs, fb, out);
}
